// ScNet_85126251806941
// MI455X (gfx1250) — 8-bit matrix operand order repaired and checked by reading, compiled, not run on hardware
//
#include <hip/hip_runtime.h>
#include <hip/hip_bf16.h>

#define L_BITS   1280
#define L_WORDS  40

typedef __attribute__((ext_vector_type(8))) int v8i;

__global__ void sc_bin_act_i8(const float* __restrict__ rm,
                              const float* __restrict__ act,
                              signed char* __restrict__ out,
                              long total, int act_mod)
{
    long i = (long)blockIdx.x * blockDim.x + threadIdx.x;
    long stride = (long)gridDim.x * blockDim.x;
    long total4 = total >> 2;
    for (long e4 = i; e4 < total4; e4 += stride) {
        long e = e4 << 2;
        float4 r = ((const float4*)rm)[e4];
        long stream = e / L_BITS;
        float t = (act[(int)(stream % act_mod)] + 1.0f) * 0.5f;
        char4 o;
        o.x = (t > r.x) ? 1 : -1;
        o.y = (t > r.y) ? 1 : -1;
        o.z = (t > r.z) ? 1 : -1;
        o.w = (t > r.w) ? 1 : -1;
        ((char4*)out)[e4] = o;
    }
}

__global__ void sc_bin_w_i8(const float* __restrict__ rm,
                            const float* __restrict__ w,
                            signed char* __restrict__ out,
                            int n_outer)
{
    long i = (long)blockIdx.x * blockDim.x + threadIdx.x;
    long stride = (long)gridDim.x * blockDim.x;
    long total4 = (long)n_outer * 32 * L_BITS / 4;
    for (long e4 = i; e4 < total4; e4 += stride) {
        long e = e4 << 2;
        long stream = e / L_BITS;
        int kpad = (int)(stream & 31);
        int outer = (int)(stream >> 5);
        char4 o;
        if (kpad < 25) {
            long srs = (long)outer * 25 + kpad;
            float t = (w[srs] + 1.0f) * 0.5f;
            float4 r = ((const float4*)rm)[srs * (L_BITS / 4) + ((e % L_BITS) >> 2)];
            o.x = (t > r.x) ? 1 : -1;
            o.y = (t > r.y) ? 1 : -1;
            o.z = (t > r.z) ? 1 : -1;
            o.w = (t > r.w) ? 1 : -1;
        } else {
            o.x = o.y = o.z = o.w = 0;
        }
        ((char4*)out)[e4] = o;
    }
}

__global__ void sc_pack_bits(const float* __restrict__ rm,
                             const float* __restrict__ vals,
                             unsigned int* __restrict__ out,
                             long total_words)
{
    int lane = threadIdx.x & 31;
    long wave = ((long)blockIdx.x * blockDim.x + threadIdx.x) >> 5;
    long nwaves = ((long)gridDim.x * blockDim.x) >> 5;
    for (long w = wave; w < total_words; w += nwaves) {
        long stream = w / L_WORDS;
        int wl = (int)(w % L_WORDS);
        float r = rm[stream * L_BITS + wl * 32 + lane];
        float t = (vals[stream] + 1.0f) * 0.5f;
        unsigned long long full = __ballot(t > r);
        if (lane == 0) out[w] = (unsigned int)full;
    }
}

__global__ void sc_conv_wmma(const signed char* __restrict__ Abuf,
                             const signed char* __restrict__ Bbuf,
                             int* __restrict__ Dbuf,
                             int Mrows, int mtiles)
{
    int bid = blockIdx.x;
    int outer = bid / mtiles;
    int mt = bid % mtiles;
    const signed char* A = Abuf + (long)outer * Mrows * L_BITS;
    const signed char* B = Bbuf + (long)outer * 32 * L_BITS;
    int* D = Dbuf + (long)outer * Mrows * 25;

    int lane = threadIdx.x;
    int half = lane >> 4;
    int l16  = lane & 15;
    int mrow = mt * 16 + l16;

    v8i acc0 = {};
    v8i acc1 = {};

    for (int ks = 0; ks < 20; ++ks) {
        int kbase = ks * 64;
        v8i afrag, bfrag0, bfrag1;
#pragma unroll
        for (int v = 0; v < 8; ++v) {
            int ka = kbase + ((v >> 2) << 5) + (((v >> 1) & 1) << 4)
                   + ((v & 1) << 2) + (half << 3);
            afrag[v] = *(const int*)(A + (long)mrow * L_BITS + ka);
            int kb = kbase + ((v >> 2) << 5) + (((v >> 1) & 1) << 4)
                   + ((v & 1) << 2) + (half << 3);
            bfrag0[v] = *(const int*)(B + (long)l16 * L_BITS + kb);
            bfrag1[v] = *(const int*)(B + (long)(16 + l16) * L_BITS + kb);
        }
        acc0 = __builtin_amdgcn_wmma_i32_16x16x64_iu8(true, afrag, true, bfrag0,
                                                      acc0, false, false);
        acc1 = __builtin_amdgcn_wmma_i32_16x16x64_iu8(true, afrag, true, bfrag1,
                                                      acc1, false, false);
    }
#pragma unroll
    for (int r = 0; r < 8; ++r) {
        int m = mt * 16 + r + half * 8;
        D[(long)m * 25 + l16] = acc0[r];
        if (16 + l16 < 25) D[(long)m * 25 + 16 + l16] = acc1[r];
    }
}

__global__ void sc_gp1(const int* __restrict__ D1, const float* __restrict__ b1,
                       float* __restrict__ p1)
{
    __shared__ float c1s[6 * 576];
    __shared__ float pools[864];
    __shared__ float rmn[256], rmx[256];
    int tid = threadIdx.x;
    for (int e = tid; e < 6 * 576; e += 256) {
        int o = e / 576, p = e % 576;
        int r = p / 24, c = p % 24;
        int s = 0;
#pragma unroll
        for (int kr = 0; kr < 5; ++kr)
#pragma unroll
            for (int kc = 0; kc < 5; ++kc) {
                int n = (r + kr) * 28 + (c + kc);
                s += D1[(o * 784 + n) * 25 + kr * 5 + kc];
            }
        float v = (float)s / (float)L_BITS + b1[o];
        c1s[e] = v > 0.0f ? v : 0.0f;
    }
    __syncthreads();
    float lmn = 1e30f, lmx = -1e30f;
    for (int e = tid; e < 864; e += 256) {
        int o = e / 144, q = e % 144;
        int r = q / 12, c = q % 12;
        const float* base = &c1s[o * 576];
        float a = base[(2 * r) * 24 + 2 * c];
        float b = base[(2 * r) * 24 + 2 * c + 1];
        float cc = base[(2 * r + 1) * 24 + 2 * c];
        float d = base[(2 * r + 1) * 24 + 2 * c + 1];
        float m = fmaxf(fmaxf(a, b), fmaxf(cc, d));
        pools[e] = m;
        lmn = fminf(lmn, m); lmx = fmaxf(lmx, m);
    }
    rmn[tid] = lmn; rmx[tid] = lmx;
    __syncthreads();
    for (int s = 128; s > 0; s >>= 1) {
        if (tid < s) {
            rmn[tid] = fminf(rmn[tid], rmn[tid + s]);
            rmx[tid] = fmaxf(rmx[tid], rmx[tid + s]);
        }
        __syncthreads();
    }
    float mn = rmn[0];
    float inv = 1.0f / (rmx[0] - mn);
    for (int e = tid; e < 864; e += 256) p1[e] = (pools[e] - mn) * inv;
}

__global__ void sc_gp2(const int* __restrict__ D2, const float* __restrict__ b2,
                       float* __restrict__ flatv)
{
    __shared__ float c2s[16 * 64];
    __shared__ float pools[256];
    __shared__ float rmn[256], rmx[256];
    int tid = threadIdx.x;
    for (int e = tid; e < 16 * 64; e += 256) {
        int o = e / 64, p = e % 64;
        int r = p / 8, c = p % 8;
        int s = 0;
        for (int i = 0; i < 6; ++i) {
            const int* Dbase = D2 + (long)(o * 6 + i) * 144 * 25;
#pragma unroll
            for (int kr = 0; kr < 5; ++kr)
#pragma unroll
                for (int kc = 0; kc < 5; ++kc) {
                    int n = (r + kr) * 12 + (c + kc);
                    s += Dbase[n * 25 + kr * 5 + kc];
                }
        }
        float v = (float)s / (float)L_BITS + b2[o];
        c2s[e] = v > 0.0f ? v : 0.0f;
    }
    __syncthreads();
    float lmn = 1e30f, lmx = -1e30f;
    if (tid < 256) {
        int o = tid / 16, q = tid % 16;
        int r = q / 4, c = q % 4;
        const float* base = &c2s[o * 64];
        float a = base[(2 * r) * 8 + 2 * c];
        float b = base[(2 * r) * 8 + 2 * c + 1];
        float cc = base[(2 * r + 1) * 8 + 2 * c];
        float d = base[(2 * r + 1) * 8 + 2 * c + 1];
        float m = fmaxf(fmaxf(a, b), fmaxf(cc, d));
        pools[tid] = m;
        lmn = m; lmx = m;
    }
    rmn[tid] = lmn; rmx[tid] = lmx;
    __syncthreads();
    for (int s = 128; s > 0; s >>= 1) {
        if (tid < s) {
            rmn[tid] = fminf(rmn[tid], rmn[tid + s]);
            rmx[tid] = fmaxf(rmx[tid], rmx[tid + s]);
        }
        __syncthreads();
    }
    float mn = rmn[0];
    float inv = 1.0f / (rmx[0] - mn);
    flatv[tid] = (pools[tid] - mn) * inv;
}

__global__ void sc_fc(const unsigned int* __restrict__ bx,
                      const unsigned int* __restrict__ bw,
                      const unsigned int* __restrict__ bb,
                      float* __restrict__ out, int N, int M)
{
    __shared__ int Dacc[128];
    int tid = threadIdx.x;
    if (tid < 128) Dacc[tid] = 0;
    __syncthreads();
    int totalPairs = M * N;
    for (int pr = tid; pr < totalPairs; pr += 256) {
        int m = pr / N, n = pr % N;
        const unsigned int* x = bx + (long)n * L_WORDS;
        const unsigned int* wv = bw + ((long)n * M + m) * L_WORDS;
        int pc = 0;
#pragma unroll
        for (int j = 0; j < L_WORDS; ++j) pc += __popc(x[j] ^ wv[j]);
        atomicAdd(&Dacc[m], pc);
    }
    __syncthreads();
    if (tid < M) {
        int D = N * L_BITS - 2 * Dacc[tid];
        const unsigned int* bbv = bb + (long)tid * L_WORDS;
        int pb = 0;
#pragma unroll
        for (int j = 0; j < L_WORDS; ++j) pb += __popc(bbv[j]);
        int SB = 2 * pb - L_BITS;
        out[tid] = (float)(D + SB) / (float)L_BITS;
    }
}

__global__ void sc_relu_norm(const float* __restrict__ in,
                             float* __restrict__ out, int M)
{
    __shared__ float rmn[128], rmx[128];
    int tid = threadIdx.x;
    float v = (tid < M) ? fmaxf(in[tid], 0.0f) : 0.0f;
    rmn[tid] = (tid < M) ? v : 1e30f;
    rmx[tid] = (tid < M) ? v : -1e30f;
    __syncthreads();
    for (int s = 64; s > 0; s >>= 1) {
        if (tid < s) {
            rmn[tid] = fminf(rmn[tid], rmn[tid + s]);
            rmx[tid] = fmaxf(rmx[tid], rmx[tid + s]);
        }
        __syncthreads();
    }
    if (tid < M) out[tid] = (v - rmn[0]) / (rmx[0] - rmn[0]);
}

static inline int grid_for(long work, int block, int cap)
{
    long g = (work + block - 1) / block;
    if (g > cap) g = cap;
    if (g < 1) g = 1;
    return (int)g;
}
static inline size_t alup(size_t x) { return (x + 255) & ~(size_t)255; }

extern "C" void kernel_launch(void* const* d_in, const int* in_sizes, int n_in,
                              void* d_out, int out_size, void* d_ws, size_t ws_size,
                              hipStream_t stream)
{
    const float* x     = (const float*)d_in[0];
    const float* w1    = (const float*)d_in[1];
    const float* b1    = (const float*)d_in[2];
    const float* w2    = (const float*)d_in[3];
    const float* b2    = (const float*)d_in[4];
    const float* wfc3  = (const float*)d_in[5];
    const float* bfc3  = (const float*)d_in[6];
    const float* wfc4  = (const float*)d_in[7];
    const float* bfc4  = (const float*)d_in[8];
    const float* wfc5  = (const float*)d_in[9];
    const float* bfc5  = (const float*)d_in[10];
    const float* rm_in1 = (const float*)d_in[11];
    const float* rm_k1  = (const float*)d_in[12];
    const float* rm_in2 = (const float*)d_in[13];
    const float* rm_k2  = (const float*)d_in[14];
    const float* rm_x3  = (const float*)d_in[15];
    const float* rm_w3  = (const float*)d_in[16];
    const float* rm_b3  = (const float*)d_in[17];
    const float* rm_x4  = (const float*)d_in[18];
    const float* rm_w4  = (const float*)d_in[19];
    const float* rm_b4  = (const float*)d_in[20];
    const float* rm_x5  = (const float*)d_in[21];
    const float* rm_w5  = (const float*)d_in[22];
    const float* rm_b5  = (const float*)d_in[23];

    char* ws = (char*)d_ws;
    size_t off = 0;
    auto take = [&](size_t bytes) { size_t o = off; off = alup(off + bytes); return o; };

    signed char* sx1 = (signed char*)(ws + take((size_t)6 * 784 * L_BITS));
    signed char* sw1 = (signed char*)(ws + take((size_t)6 * 32 * L_BITS));
    int*         D1  = (int*)(ws + take((size_t)6 * 784 * 25 * 4));
    float*       p1v = (float*)(ws + take((size_t)864 * 4));
    signed char* sx2 = (signed char*)(ws + take((size_t)96 * 144 * L_BITS));
    signed char* sw2 = (signed char*)(ws + take((size_t)96 * 32 * L_BITS));
    int*         D2  = (int*)(ws + take((size_t)96 * 144 * 25 * 4));
    float*       flv = (float*)(ws + take((size_t)256 * 4));
    unsigned int* bx3 = (unsigned int*)(ws + take((size_t)256 * L_WORDS * 4));
    unsigned int* bw3 = (unsigned int*)(ws + take((size_t)256 * 120 * L_WORDS * 4));
    unsigned int* bb3 = (unsigned int*)(ws + take((size_t)120 * L_WORDS * 4));
    float*       raw3 = (float*)(ws + take(512));
    float*       f1   = (float*)(ws + take(512));
    unsigned int* bx4 = (unsigned int*)(ws + take((size_t)120 * L_WORDS * 4));
    unsigned int* bw4 = (unsigned int*)(ws + take((size_t)120 * 84 * L_WORDS * 4));
    unsigned int* bb4 = (unsigned int*)(ws + take((size_t)84 * L_WORDS * 4));
    float*       raw4 = (float*)(ws + take(512));
    float*       f2   = (float*)(ws + take(512));
    unsigned int* bx5 = (unsigned int*)(ws + take((size_t)84 * L_WORDS * 4));
    unsigned int* bw5 = (unsigned int*)(ws + take((size_t)84 * 10 * L_WORDS * 4));
    unsigned int* bb5 = (unsigned int*)(ws + take((size_t)10 * L_WORDS * 4));
    (void)ws_size; (void)in_sizes; (void)n_in; (void)out_size;

    const int BLK = 256;
    const int CAP = 8192;

    {
        long t = (long)6 * 784 * L_BITS;
        sc_bin_act_i8<<<grid_for(t / 4, BLK, CAP), BLK, 0, stream>>>(rm_in1, x, sx1, t, 784);
    }
    sc_bin_w_i8<<<grid_for((long)6 * 32 * L_BITS / 4, BLK, CAP), BLK, 0, stream>>>(rm_k1, w1, sw1, 6);
    sc_bin_w_i8<<<grid_for((long)96 * 32 * L_BITS / 4, BLK, CAP), BLK, 0, stream>>>(rm_k2, w2, sw2, 96);
    {
        long tw = (long)256 * 120 * L_WORDS;
        sc_pack_bits<<<grid_for(tw * 32, BLK, CAP), BLK, 0, stream>>>(rm_w3, wfc3, bw3, tw);
        tw = (long)120 * L_WORDS;
        sc_pack_bits<<<grid_for(tw * 32, BLK, CAP), BLK, 0, stream>>>(rm_b3, bfc3, bb3, tw);
        tw = (long)120 * 84 * L_WORDS;
        sc_pack_bits<<<grid_for(tw * 32, BLK, CAP), BLK, 0, stream>>>(rm_w4, wfc4, bw4, tw);
        tw = (long)84 * L_WORDS;
        sc_pack_bits<<<grid_for(tw * 32, BLK, CAP), BLK, 0, stream>>>(rm_b4, bfc4, bb4, tw);
        tw = (long)84 * 10 * L_WORDS;
        sc_pack_bits<<<grid_for(tw * 32, BLK, CAP), BLK, 0, stream>>>(rm_w5, wfc5, bw5, tw);
        tw = (long)10 * L_WORDS;
        sc_pack_bits<<<grid_for(tw * 32, BLK, CAP), BLK, 0, stream>>>(rm_b5, bfc5, bb5, tw);
    }

    sc_conv_wmma<<<6 * 49, 32, 0, stream>>>(sx1, sw1, D1, 784, 49);
    sc_gp1<<<1, 256, 0, stream>>>(D1, b1, p1v);

    {
        long t = (long)96 * 144 * L_BITS;
        sc_bin_act_i8<<<grid_for(t / 4, BLK, CAP), BLK, 0, stream>>>(rm_in2, p1v, sx2, t, 864);
    }
    sc_conv_wmma<<<96 * 9, 32, 0, stream>>>(sx2, sw2, D2, 144, 9);
    sc_gp2<<<1, 256, 0, stream>>>(D2, b2, flv);

    sc_pack_bits<<<grid_for((long)256 * L_WORDS * 32, BLK, CAP), BLK, 0, stream>>>(rm_x3, flv, bx3, (long)256 * L_WORDS);
    sc_fc<<<1, 256, 0, stream>>>(bx3, bw3, bb3, raw3, 256, 120);
    sc_relu_norm<<<1, 128, 0, stream>>>(raw3, f1, 120);

    sc_pack_bits<<<grid_for((long)120 * L_WORDS * 32, BLK, CAP), BLK, 0, stream>>>(rm_x4, f1, bx4, (long)120 * L_WORDS);
    sc_fc<<<1, 256, 0, stream>>>(bx4, bw4, bb4, raw4, 120, 84);
    sc_relu_norm<<<1, 128, 0, stream>>>(raw4, f2, 84);

    sc_pack_bits<<<grid_for((long)84 * L_WORDS * 32, BLK, CAP), BLK, 0, stream>>>(rm_x5, f2, bx5, (long)84 * L_WORDS);
    sc_fc<<<1, 256, 0, stream>>>(bx5, bw5, bb5, (float*)d_out, 84, 10);
}
